// CausalHolographicQKV_35570919146257
// MI455X (gfx1250) — compile-verified, matrix operand order checked by reading, not run on hardware
//
#include <hip/hip_runtime.h>
#include <math.h>

// ---------------------------------------------------------------------------
// Problem constants (reference: B=4, S=4096, D=1024)
// ---------------------------------------------------------------------------
#define NB 4
#define NS 4096
#define ND 1024
static constexpr size_t BS_  = (size_t)NB * NS;   // 16384 tokens
static constexpr size_t BSD_ = BS_ * ND;          // 16,777,216 elements

// ---------------------------------------------------------------------------
// WMMA vector types (gfx1250, wave32)
// ---------------------------------------------------------------------------
typedef __attribute__((ext_vector_type(8)))  __bf16 bhalf8;
typedef __attribute__((ext_vector_type(16))) __bf16 bhalf16;
typedef __attribute__((ext_vector_type(8)))  float  v8f;

// ---------------------------------------------------------------------------
// Async global->LDS copy (CDNA5 GLOBAL_LOAD_ASYNC_TO_LDS_B128, ASYNCcnt).
// Toolchain signature (from probe diagnostics): the b128 builtin takes
// pointers to 16-byte int vectors in AS1 (global) and AS3 (LDS).
// ---------------------------------------------------------------------------
#if defined(__has_builtin)
# if __has_builtin(__builtin_amdgcn_global_load_async_to_lds_b128) && \
     __has_builtin(__builtin_amdgcn_s_wait_asynccnt)
#  define USE_ASYNC 1
# endif
#endif
#ifndef USE_ASYNC
# define USE_ASYNC 0
#endif

#if USE_ASYNC
typedef int v4i_gcc __attribute__((vector_size(16)));
typedef __attribute__((address_space(1))) v4i_gcc* gptr128_t;   // global v4i*
typedef __attribute__((address_space(3))) v4i_gcc* lptr128_t;   // LDS    v4i*
__device__ __forceinline__ void async_cp16(__bf16* l, const __bf16* g) {
    __builtin_amdgcn_global_load_async_to_lds_b128(
        (gptr128_t)(void*)g, (lptr128_t)(void*)l, 0, 0);
}
# define WAIT_ASYNC(N) __builtin_amdgcn_s_wait_asynccnt(N)
#else
__device__ __forceinline__ void async_cp16(__bf16* l, const __bf16* g) {
    *(bhalf8*)l = *(const bhalf8*)g;    // synchronous fallback
}
# define WAIT_ASYNC(N)
#endif

// ---------------------------------------------------------------------------
// Complex helpers
// ---------------------------------------------------------------------------
__device__ __forceinline__ float2 cmul(float2 a, float2 b) {
    return make_float2(a.x * b.x - a.y * b.y, a.x * b.y + a.y * b.x);
}

// ---------------------------------------------------------------------------
// fp32 -> bf16 staging kernel (halves GEMM global-read bytes).
// Each thread converts 8 contiguous floats -> one 16B bf16 store.
// ---------------------------------------------------------------------------
__global__ __launch_bounds__(256) void cvt_bf16_kernel(
    const float* __restrict__ src, __bf16* __restrict__ dst, int n8)
{
    const int i = blockIdx.x * 256 + threadIdx.x;
    if (i >= n8) return;
    const float4* s = (const float4*)src + (size_t)i * 2;
    float4 a = s[0], b = s[1];
    bhalf8 v;
    v[0] = (__bf16)a.x; v[1] = (__bf16)a.y; v[2] = (__bf16)a.z; v[3] = (__bf16)a.w;
    v[4] = (__bf16)b.x; v[5] = (__bf16)b.y; v[6] = (__bf16)b.z; v[7] = (__bf16)b.w;
    *(bhalf8*)(dst + (size_t)i * 8) = v;
}

// ---------------------------------------------------------------------------
// GEMM:  lin[which][m][n] = sum_k x[m][k] * W[which][n][k] + bias[which][n]
// Tile 128x128x32, 256 threads = 8 wave32 waves (each 32x64 = 2x4 WMMA accs),
// double-buffered LDS fed by async global->LDS copies (4 b128 per wave/tile).
// ---------------------------------------------------------------------------
#define TM 128
#define TN 128
#define TK 32
#define LDP (TK + 8)   // LDS row pitch in halves (80 B, keeps 16B alignment)

// Per-lane fragment per CDNA5 16-bit A/B layout: lane l -> row (l&15);
// halves e=0..7 -> K=(l>>4)*8+e ; e=8..15 -> K=16+(l>>4)*8+(e-8).
__device__ __forceinline__ bhalf16 load_frag(const __bf16* rowPtr, int krow) {
    bhalf8 lo = *(const bhalf8*)(rowPtr + krow * 8);
    bhalf8 hi = *(const bhalf8*)(rowPtr + 16 + krow * 8);
    return __builtin_shufflevector(lo, hi, 0, 1, 2, 3, 4, 5, 6, 7,
                                           8, 9, 10, 11, 12, 13, 14, 15);
}

__global__ __launch_bounds__(256) void gemm_qkv_kernel(
    const __bf16* __restrict__ xb, const __bf16* __restrict__ wb,
    const float* __restrict__ bq, const float* __restrict__ bk,
    const float* __restrict__ bv, float* __restrict__ lin)
{
    const int which = blockIdx.z;
    const __bf16* __restrict__ W = wb + (size_t)which * ND * ND;
    const float* __restrict__ bias = (which == 0) ? bq : ((which == 1) ? bk : bv);
    float* __restrict__ out = lin + (size_t)which * BSD_;

    __shared__ __align__(16) __bf16 As[2][TM][LDP];
    __shared__ __align__(16) __bf16 Bs[2][TN][LDP];

    const int tid  = threadIdx.x;
    const int lane = tid & 31;
    const int wave = tid >> 5;
    const int wm   = wave & 3;   // wave M offset = wm*32
    const int wn   = wave >> 2;  // wave N offset = wn*64

    const int blockM = blockIdx.y * TM;
    const int blockN = blockIdx.x * TN;

    v8f acc[2][4];
#pragma unroll
    for (int mi = 0; mi < 2; ++mi)
#pragma unroll
        for (int ni = 0; ni < 4; ++ni)
            acc[mi][ni] = (v8f){0.f, 0.f, 0.f, 0.f, 0.f, 0.f, 0.f, 0.f};

    // 16B-chunk mapping: tile = 128 rows x 4 chunks (8 halves each) = 512
    // chunks per matrix; 256 threads -> 2 chunks each for A and for B.
    const int c0 = tid,        r0 = c0 >> 2, q0 = (c0 & 3) * 8;
    const int c1 = tid + 256,  r1 = c1 >> 2, q1 = (c1 & 3) * 8;

    const int krow = lane >> 4;  // fragment K half select
    const int frow = lane & 15;  // fragment row-in-tile

    auto issue_tile = [&](int kk, int buf) {
        async_cp16(&As[buf][r0][q0], xb + (size_t)(blockM + r0) * ND + kk + q0);
        async_cp16(&As[buf][r1][q1], xb + (size_t)(blockM + r1) * ND + kk + q1);
        async_cp16(&Bs[buf][r0][q0], W  + (size_t)(blockN + r0) * ND + kk + q0);
        async_cp16(&Bs[buf][r1][q1], W  + (size_t)(blockN + r1) * ND + kk + q1);
    };

    constexpr int NT = ND / TK;      // 32 K-tiles
    issue_tile(0, 0);

    for (int t = 0; t < NT; ++t) {
        const int buf = t & 1;
        if (t + 1 < NT) {
            issue_tile((t + 1) * TK, buf ^ 1);
            WAIT_ASYNC(4);           // retire tile t (in-order ASYNCcnt)
        } else {
            WAIT_ASYNC(0);
        }
        __syncthreads();             // tile t visible to all waves

        bhalf16 afrag[2], bfrag[4];
#pragma unroll
        for (int mi = 0; mi < 2; ++mi)
            afrag[mi] = load_frag(&As[buf][wm * 32 + mi * 16 + frow][0], krow);
#pragma unroll
        for (int ni = 0; ni < 4; ++ni)
            bfrag[ni] = load_frag(&Bs[buf][wn * 64 + ni * 16 + frow][0], krow);

#pragma unroll
        for (int mi = 0; mi < 2; ++mi)
#pragma unroll
            for (int ni = 0; ni < 4; ++ni)
                acc[mi][ni] = __builtin_amdgcn_wmma_f32_16x16x32_bf16(
                    false, afrag[mi], false, bfrag[ni],
                    (short)0, acc[mi][ni], false, false);
        __syncthreads();             // all reads of buf done before its refill
    }

    // Epilogue: C/D layout: VGPR r -> row (lane>>4)*8 + r, col lane&15
#pragma unroll
    for (int mi = 0; mi < 2; ++mi) {
#pragma unroll
        for (int ni = 0; ni < 4; ++ni) {
            const int n   = blockN + wn * 64 + ni * 16 + frow;
            const float bb = bias[n];
#pragma unroll
            for (int r = 0; r < 8; ++r) {
                const int m = blockM + wm * 32 + mi * 16 + (lane >> 4) * 8 + r;
                out[(size_t)m * ND + n] = acc[mi][ni][r] + bb;
            }
        }
    }
}

// ---------------------------------------------------------------------------
// 1024-point radix-2 complex FFT in LDS, 256 threads per row.
// ---------------------------------------------------------------------------
__device__ void fft1024(float2* sh, int tid, bool inverse)
{
    for (int i = tid; i < 1024; i += 256) {          // bit-reversal (10 bits)
        int j = (int)(__brev((unsigned)i) >> 22);
        if (j > i) { float2 t = sh[i]; sh[i] = sh[j]; sh[j] = t; }
    }
    __syncthreads();

    const float sgn = inverse ? 1.0f : -1.0f;
#pragma unroll
    for (int slog = 1; slog <= 10; ++slog) {
        const int len  = 1 << slog;
        const int half = len >> 1;
        for (int i = tid; i < 512; i += 256) {
            const int blk = i >> (slog - 1);
            const int off = i & (half - 1);
            const int pos = blk * len + off;
            const float ang = sgn * 6.28318530717958647692f * (float)off / (float)len;
            float s, c;
            __sincosf(ang, &s, &c);
            const float2 w = make_float2(c, s);
            const float2 u = sh[pos];
            const float2 t = cmul(w, sh[pos + half]);
            sh[pos]        = make_float2(u.x + t.x, u.y + t.y);
            sh[pos + half] = make_float2(u.x - t.x, u.y - t.y);
        }
        __syncthreads();
    }
}

// ---------------------------------------------------------------------------
// Per-token spectral kernel: FFT(K),FFT(V),FFT(Q); unit-magnitude normalize;
// Fkv = Fk*Fv  and  Fqc = conj(Fq).
// ---------------------------------------------------------------------------
__global__ __launch_bounds__(256) void spectral_qkv_kernel(
    const float* __restrict__ lin,
    float2* __restrict__ fkv, float2* __restrict__ fqc)
{
    __shared__ float2 shA[1024];
    __shared__ float2 shK[1024];
    const int tid = threadIdx.x;
    const size_t row = blockIdx.x;
    const float* __restrict__ qrow = lin + row * ND;
    const float* __restrict__ krow = lin + BSD_ + row * ND;
    const float* __restrict__ vrow = lin + 2 * BSD_ + row * ND;

    if (tid < 32) {   // pull V and Q rows toward the caches while K FFTs
        __builtin_prefetch(vrow + tid * 32, 0, 1);
        __builtin_prefetch(qrow + tid * 32, 0, 1);
    }

    // ---- K
    for (int i = tid; i < 1024; i += 256) shA[i] = make_float2(krow[i], 0.f);
    __syncthreads();
    fft1024(shA, tid, false);
    for (int i = tid; i < 1024; i += 256) {
        float2 f = shA[i];
        float inv = 1.f / (sqrtf(f.x * f.x + f.y * f.y) + 1e-8f);
        shK[i] = make_float2(f.x * inv, f.y * inv);
    }
    // ---- V
    for (int i = tid; i < 1024; i += 256) shA[i] = make_float2(vrow[i], 0.f);
    __syncthreads();
    fft1024(shA, tid, false);
    for (int i = tid; i < 1024; i += 256) {
        float2 f = shA[i];
        float inv = 1.f / (sqrtf(f.x * f.x + f.y * f.y) + 1e-8f);
        float2 fv = make_float2(f.x * inv, f.y * inv);
        fkv[row * ND + i] = cmul(shK[i], fv);
    }
    // ---- Q
    for (int i = tid; i < 1024; i += 256) shA[i] = make_float2(qrow[i], 0.f);
    __syncthreads();
    fft1024(shA, tid, false);
    for (int i = tid; i < 1024; i += 256) {
        float2 f = shA[i];
        float inv = 1.f / (sqrtf(f.x * f.x + f.y * f.y) + 1e-8f);
        fqc[row * ND + i] = make_float2(f.x * inv, -f.y * inv);
    }
}

// ---------------------------------------------------------------------------
// Causal prefix scan of Fkv over S (frequency domain, complex), two levels.
// ---------------------------------------------------------------------------
__global__ __launch_bounds__(256) void scan_pass1_kernel(
    float2* __restrict__ fkv, float2* __restrict__ bsum)
{
    const int d  = blockIdx.x * 256 + threadIdx.x;
    const int sb = blockIdx.y;
    const int b  = blockIdx.z;
    size_t base = ((size_t)b * NS + (size_t)sb * 256) * ND + d;
    float2 acc = make_float2(0.f, 0.f);
    for (int s = 0; s < 256; ++s) {
        size_t idx = base + (size_t)s * ND;
        float2 v = fkv[idx];
        acc.x += v.x; acc.y += v.y;
        fkv[idx] = acc;
    }
    bsum[((size_t)(b * 16 + sb)) * ND + d] = acc;
}

__global__ __launch_bounds__(256) void scan_pass2_kernel(float2* __restrict__ bsum)
{
    const int d = blockIdx.x * 256 + threadIdx.x;
    const int b = blockIdx.z;
    float2 acc = make_float2(0.f, 0.f);
#pragma unroll
    for (int blk = 0; blk < 16; ++blk) {
        size_t idx = ((size_t)(b * 16 + blk)) * ND + d;
        float2 t = bsum[idx];
        bsum[idx] = acc;           // exclusive offset for this block
        acc.x += t.x; acc.y += t.y;
    }
}

// ---------------------------------------------------------------------------
// Output: F = (partial + blockOffset) * conj(Fq);  out = ifft(F).real / N
// ---------------------------------------------------------------------------
__global__ __launch_bounds__(256) void unbind_out_kernel(
    const float2* __restrict__ fkv, const float2* __restrict__ boff,
    const float2* __restrict__ fqc, float* __restrict__ out)
{
    __shared__ float2 sh[1024];
    const int tid = threadIdx.x;
    const size_t row = blockIdx.x;
    const int b   = (int)(row / NS);
    const int s   = (int)(row % NS);
    const int blk = s >> 8;

    for (int i = tid; i < 1024; i += 256) {
        float2 v = fkv[row * ND + i];
        float2 o = boff[((size_t)(b * 16 + blk)) * ND + i];
        float2 t = make_float2(v.x + o.x, v.y + o.y);
        sh[i] = cmul(t, fqc[row * ND + i]);
    }
    __syncthreads();
    fft1024(sh, tid, true);
    const float scale = 1.0f / 1024.0f;
    for (int i = tid; i < 1024; i += 256)
        out[row * ND + i] = sh[i].x * scale;
}

// ---------------------------------------------------------------------------
// Launch
// ---------------------------------------------------------------------------
extern "C" void kernel_launch(void* const* d_in, const int* in_sizes, int n_in,
                              void* d_out, int out_size, void* d_ws, size_t ws_size,
                              hipStream_t stream)
{
    const float* x  = (const float*)d_in[0];
    const float* Wq = (const float*)d_in[1];
    const float* bq = (const float*)d_in[2];
    const float* Wk = (const float*)d_in[3];
    const float* bk = (const float*)d_in[4];
    const float* Wv = (const float*)d_in[5];
    const float* bv = (const float*)d_in[6];
    float* out = (float*)d_out;

    // Workspace layout:
    //   xb  : BSD bf16          (x staged as bf16)            32 MB
    //   wb  : 3*ND*ND bf16      (Wq|Wk|Wv staged as bf16)      6 MB
    //   lin : 3*BSD fp32        (Q,K,V linear outputs)       192 MB
    //   fkv : BSD float2        (spectra product / scanned)  128 MB
    //   fqc : BSD float2        (conj(Fq))                   128 MB
    //   bsum: NB*16*ND float2   (scan block offsets)         0.5 MB
    char* ws = (char*)d_ws;
    __bf16* xb = (__bf16*)ws;
    __bf16* wb = xb + BSD_;
    float*  lin  = (float*)(wb + (size_t)3 * ND * ND);
    float2* fkv  = (float2*)((char*)lin + 3 * BSD_ * sizeof(float));
    float2* fqc  = fkv + BSD_;
    float2* bsum = fqc + BSD_;

    // Stage operands as bf16 (one-time ~100 MB stream; halves GEMM reads).
    {
        const int n8x = (int)(BSD_ / 8);
        const int n8w = (ND * ND) / 8;
        cvt_bf16_kernel<<<dim3((n8x + 255) / 256), 256, 0, stream>>>(x,  xb, n8x);
        cvt_bf16_kernel<<<dim3((n8w + 255) / 256), 256, 0, stream>>>(Wq, wb + (size_t)0 * ND * ND, n8w);
        cvt_bf16_kernel<<<dim3((n8w + 255) / 256), 256, 0, stream>>>(Wk, wb + (size_t)1 * ND * ND, n8w);
        cvt_bf16_kernel<<<dim3((n8w + 255) / 256), 256, 0, stream>>>(Wv, wb + (size_t)2 * ND * ND, n8w);
    }

    gemm_qkv_kernel<<<dim3(ND / TN, (unsigned)(BS_ / TM), 3), 256, 0, stream>>>(
        xb, wb, bq, bk, bv, lin);

    spectral_qkv_kernel<<<dim3((unsigned)BS_), 256, 0, stream>>>(lin, fkv, fqc);

    scan_pass1_kernel<<<dim3(ND / 256, 16, NB), 256, 0, stream>>>(fkv, bsum);
    scan_pass2_kernel<<<dim3(ND / 256, 1, NB), 256, 0, stream>>>(bsum);

    unbind_out_kernel<<<dim3((unsigned)BS_), 256, 0, stream>>>(fkv, bsum, fqc, out);
}
